// TestAllReduceRMSNormModel_45200235823678
// MI455X (gfx1250) — matrix operand order repaired and checked by reading, not compiled, not run on hardware
//
#include <hip/hip_runtime.h>
#include <stdint.h>

// ---------------------------------------------------------------------------
// AllReduce(identity)+RMSNorm+3xGEMM chain for MI455X (gfx1250, wave32, WMMA)
// fp32 in/out, bf16 tensor-core GEMMs (v_wmma_f32_16x16x32_bf16), fp32 accum.
// Global->LDS staging uses CDNA5 async copies (ASYNCcnt / s_wait_asynccnt).
// ---------------------------------------------------------------------------

typedef unsigned short ushort_t;
typedef __attribute__((ext_vector_type(16))) __bf16 v16bf;
typedef __attribute__((ext_vector_type(8)))  float  v8f;

constexpr int Hdim = 4096;   // hidden size
constexpr int Tdim = 8192;   // tokens

// round-to-nearest-even f32 -> bf16
__device__ __forceinline__ unsigned short f2bf(float f) {
  unsigned int u = __float_as_uint(f);
  unsigned int r = u + 0x7fffu + ((u >> 16) & 1u);
  return (unsigned short)(r >> 16);
}

// CDNA5 async global->LDS copy, 16B per lane. lds_off is an absolute LDS byte
// offset (dynamic-LDS region starts at 0 when the kernel has no static LDS).
__device__ __forceinline__ void async_load16(unsigned lds_off, const void* g) {
  asm volatile("global_load_async_to_lds_b128 %0, %1, off"
               :
               : "v"(lds_off), "v"((unsigned long long)(uintptr_t)g)
               : "memory");
}

__device__ __forceinline__ void wait_async0() {
#if __has_builtin(__builtin_amdgcn_s_wait_asynccnt)
  __builtin_amdgcn_s_wait_asynccnt(0);
#else
  asm volatile("s_wait_asynccnt 0x0" ::: "memory");
#endif
}

// ---------------------------------------------------------------------------
// Weight convert + transpose: w[K][N] f32 -> wt[N][K] bf16 (LDS-tiled)
// ---------------------------------------------------------------------------
__global__ __launch_bounds__(256)
void wconv_kernel(const float* __restrict__ w, ushort_t* __restrict__ wt) {
  __shared__ float tile[32][33];
  const int k0 = blockIdx.x * 32;
  const int n0 = blockIdx.y * 32;
  const int tx = threadIdx.x & 31;
  const int ty = threadIdx.x >> 5;   // 0..7
#pragma unroll
  for (int i = 0; i < 32; i += 8)
    tile[ty + i][tx] = w[(size_t)(k0 + ty + i) * Hdim + n0 + tx];
  __syncthreads();
#pragma unroll
  for (int i = 0; i < 32; i += 8)
    wt[(size_t)(n0 + ty + i) * Hdim + k0 + tx] = f2bf(tile[tx][ty + i]);
}

// ---------------------------------------------------------------------------
// pre: z = relu(x); resid = z; A = bf16( rmsnorm(z, g0) )
// one block (256 thr) per row of H=4096 -> 16 elems/thread
// ---------------------------------------------------------------------------
__global__ __launch_bounds__(256)
void pre_kernel(const float* __restrict__ x, const float* __restrict__ g,
                float* __restrict__ resid, ushort_t* __restrict__ A) {
  __shared__ float red[256];
  const int row = blockIdx.x;
  const int t = threadIdx.x;
  const float* xr = x + (size_t)row * Hdim;
  float v[16];
  float ss = 0.f;
#pragma unroll
  for (int i = 0; i < 16; ++i) {
    float z = xr[t + i * 256];
    z = z > 0.f ? z : 0.f;
    v[i] = z;
    ss += z * z;
  }
#pragma unroll
  for (int i = 0; i < 16; ++i) resid[(size_t)row * Hdim + t + i * 256] = v[i];
  red[t] = ss;
  __syncthreads();
  for (int w = 128; w > 0; w >>= 1) {
    if (t < w) red[t] += red[t + w];
    __syncthreads();
  }
  const float scale = rsqrtf(red[0] * (1.0f / Hdim) + 1e-6f);
#pragma unroll
  for (int i = 0; i < 16; ++i) {
    const float y = v[i] * scale * g[t + i * 256];
    A[(size_t)row * Hdim + t + i * 256] = f2bf(y);
  }
}

// ---------------------------------------------------------------------------
// addnorm: y = rmsnorm(resid, g); FINAL ? out(f32)=y : A(bf16)=y
// (resid already contains x_i + resid from the GEMM epilogue)
// ---------------------------------------------------------------------------
template <bool FINAL>
__global__ __launch_bounds__(256)
void addnorm_kernel(const float* __restrict__ resid, const float* __restrict__ g,
                    ushort_t* __restrict__ A, float* __restrict__ out) {
  __shared__ float red[256];
  const int row = blockIdx.x;
  const int t = threadIdx.x;
  const float* rr = resid + (size_t)row * Hdim;
  float v[16];
  float ss = 0.f;
#pragma unroll
  for (int i = 0; i < 16; ++i) {
    v[i] = rr[t + i * 256];
    ss += v[i] * v[i];
  }
  red[t] = ss;
  __syncthreads();
  for (int w = 128; w > 0; w >>= 1) {
    if (t < w) red[t] += red[t + w];
    __syncthreads();
  }
  const float scale = rsqrtf(red[0] * (1.0f / Hdim) + 1e-6f);
#pragma unroll
  for (int i = 0; i < 16; ++i) {
    const float y = v[i] * scale * g[t + i * 256];
    if constexpr (FINAL)
      out[(size_t)row * Hdim + t + i * 256] = y;
    else
      A[(size_t)row * Hdim + t + i * 256] = f2bf(y);
  }
}

// ---------------------------------------------------------------------------
// GEMM + residual add: resid[M,N] += A[M,K](bf16) * Wt[N,K](bf16, transposed)
// 128x128 block tile, 8 wave32 waves (4x2), wave tile 32x64 -> 8 WMMA/K-step.
// K staged in 32-wide slices, double-buffered dynamic LDS filled by
// global_load_async_to_lds_b128; padded 80B rows for bank-friendly frag reads.
// Stage loop is kept rolled (#pragma unroll 1) so each accumulator stays in a
// fixed VGPR tuple (no v_mov shuffles between ping-pong bodies).
// ---------------------------------------------------------------------------
union FragU {
  v16bf v;
  uint4 q[2];
};

constexpr int GKB = 32;                       // K per stage
constexpr int GLDA = GKB + 8;                 // 40 ushorts = 80B row stride
constexpr int BUF_USH = 128 * GLDA;           // 5120 ushorts per buffer
constexpr int BUF_BYTES = BUF_USH * 2;        // 10240 bytes per buffer
constexpr int GEMM_LDS_BYTES = 4 * BUF_BYTES; // A0|A1|B0|B1 = 40960 bytes

extern __shared__ __align__(16) ushort_t smem[];

__global__ __launch_bounds__(256)
void gemm_resid_kernel(const ushort_t* __restrict__ Abf,
                       const ushort_t* __restrict__ Wt,
                       float* __restrict__ resid) {
  const int t = threadIdx.x;
  const int lane = t & 31;
  const int wv = t >> 5;
  const int wm = (wv & 3) * 32;   // wave M offset in tile
  const int wn = (wv >> 2) * 64;  // wave N offset in tile
  const int l16 = lane & 15;
  const int hsel = lane >> 4;     // lane half select

  const int m0 = blockIdx.y * 128;
  const int n0 = blockIdx.x * 128;

  // cooperative load mapping: 2 threads per row, 32B contiguous per thread
  const int rowL = t >> 1;         // 0..127
  const int cL = (t & 1) * 16;     // ushort offset 0 / 16

  const ushort_t* gA = Abf + (size_t)(m0 + rowL) * Hdim + cL;
  const ushort_t* gB = Wt + (size_t)(n0 + rowL) * Hdim + cL;

  // absolute LDS byte offsets for this thread's async destinations
  const unsigned ldsA = (unsigned)(rowL * (GLDA * 2) + cL * 2);
  const unsigned ldsB = (unsigned)(2 * BUF_BYTES + rowL * (GLDA * 2) + cL * 2);

  // per-thread fragment base offsets (ushort units) within a buffer
  const int fraA = (wm + l16) * GLDA;             // + mt*16*GLDA
  const int fraB = (wn + l16) * GLDA + hsel * 16; // + nt*16*GLDA
  const int ka0 = hsel ? 16 : 0;
  const int ka1 = hsel ? 24 : 8;

  v8f acc[2][4] = {};

  // prologue: async-stage slice 0 into buffer 0
  async_load16(ldsA, gA);
  async_load16(ldsA + 16, gA + 8);
  async_load16(ldsB, gB);
  async_load16(ldsB + 16, gB + 8);

  const ushort_t* gAs = gA + GKB;  // next slice to prefetch
  const ushort_t* gBs = gB + GKB;
  unsigned cur = 0;                // current buffer byte offset (0 / BUF_BYTES)

  const int nK = Hdim / GKB;  // 128 stages
#pragma unroll 1
  for (int s = 0; s < nK; ++s) {
    wait_async0();       // this wave's stage-s async copies have landed in LDS
    __syncthreads();     // ... and everyone else's too

    const unsigned nxt = cur ^ (unsigned)BUF_BYTES;
    if (s + 1 < nK) {    // kick off next slice into the other buffer
      async_load16(ldsA + nxt, gAs);
      async_load16(ldsA + nxt + 16, gAs + 8);
      async_load16(ldsB + nxt, gBs);
      async_load16(ldsB + nxt + 16, gBs + 8);
      gAs += GKB;
      gBs += GKB;
    }

    const ushort_t* bufA =
        (const ushort_t*)((const char*)smem + cur) + fraA;
    const ushort_t* bufB =
        (const ushort_t*)((const char*)smem + 2 * BUF_BYTES + cur) + fraB;

    // A fragments (16x32 bf16): lanes 0-15 hold K 0-15, lanes 16-31
    // hold K 16-31 (matching B) -> two 16B LDS reads per lane.
    FragU af[2];
#pragma unroll
    for (int mt = 0; mt < 2; ++mt) {
      const ushort_t* r = bufA + mt * 16 * GLDA;
      af[mt].q[0] = *(const uint4*)(r + ka0);
      af[mt].q[1] = *(const uint4*)(r + ka1);
    }

    // B fragments (32x16 bf16): lane n holds K 0-15 (lanes 0-15) or K 16-31
    // (lanes 16-31) contiguously -> one 32B LDS read per lane (Wt transposed).
    FragU bfb[4];
#pragma unroll
    for (int nt = 0; nt < 4; ++nt) {
      const ushort_t* r = bufB + nt * 16 * GLDA;
      bfb[nt].q[0] = *(const uint4*)(r);
      bfb[nt].q[1] = *(const uint4*)(r + 8);
    }

#pragma unroll
    for (int mt = 0; mt < 2; ++mt)
#pragma unroll
      for (int nt = 0; nt < 4; ++nt)
        acc[mt][nt] = __builtin_amdgcn_wmma_f32_16x16x32_bf16(
            false, af[mt].v, false, bfb[nt].v, (short)0, acc[mt][nt],
            false, false);

    cur = nxt;
  }

  // epilogue: resid += acc  (C/D layout: VGPR r, lanes 0-15 -> M=r,
  // lanes 16-31 -> M=r+8, N = lane%16). Each element owned by one lane.
#pragma unroll
  for (int mt = 0; mt < 2; ++mt) {
#pragma unroll
    for (int nt = 0; nt < 4; ++nt) {
      const int col = n0 + wn + nt * 16 + l16;
#pragma unroll
      for (int r = 0; r < 8; ++r) {
        const int rowg = m0 + wm + mt * 16 + r + 8 * hsel;
        const size_t idx = (size_t)rowg * Hdim + col;
        resid[idx] += acc[mt][nt][r];
      }
    }
  }
}

// ---------------------------------------------------------------------------
// launch
// ---------------------------------------------------------------------------
extern "C" void kernel_launch(void* const* d_in, const int* in_sizes, int n_in,
                              void* d_out, int out_size, void* d_ws,
                              size_t ws_size, hipStream_t stream) {
  (void)in_sizes; (void)n_in; (void)out_size; (void)ws_size;

  const float* x  = (const float*)d_in[0];
  const float* w0 = (const float*)d_in[1];
  const float* w1 = (const float*)d_in[2];
  const float* w2 = (const float*)d_in[3];
  const float* g0 = (const float*)d_in[4];
  const float* g1 = (const float*)d_in[5];
  const float* g2 = (const float*)d_in[6];
  const float* g3 = (const float*)d_in[7];
  float* out = (float*)d_out;

  // workspace layout: resid f32 (128MB) | A bf16 (64MB) | Wt0/1/2 bf16 (96MB)
  char* ws = (char*)d_ws;
  float*    resid = (float*)ws;
  ushort_t* Abf   = (ushort_t*)(ws + (size_t)Tdim * Hdim * 4);
  ushort_t* Wt0   = (ushort_t*)(ws + (size_t)Tdim * Hdim * 4 +
                                (size_t)Tdim * Hdim * 2);
  ushort_t* Wt1 = Wt0 + (size_t)Hdim * Hdim;
  ushort_t* Wt2 = Wt1 + (size_t)Hdim * Hdim;

  dim3 tgrid(Hdim / 32, Hdim / 32);
  wconv_kernel<<<tgrid, 256, 0, stream>>>(w0, Wt0);
  wconv_kernel<<<tgrid, 256, 0, stream>>>(w1, Wt1);
  wconv_kernel<<<tgrid, 256, 0, stream>>>(w2, Wt2);

  pre_kernel<<<Tdim, 256, 0, stream>>>(x, g0, resid, Abf);

  dim3 ggrid(Hdim / 128, Tdim / 128);  // (32, 64)
  gemm_resid_kernel<<<ggrid, 256, GEMM_LDS_BYTES, stream>>>(Abf, Wt0, resid);
  addnorm_kernel<false><<<Tdim, 256, 0, stream>>>(resid, g1, Abf, nullptr);
  gemm_resid_kernel<<<ggrid, 256, GEMM_LDS_BYTES, stream>>>(Abf, Wt1, resid);
  addnorm_kernel<false><<<Tdim, 256, 0, stream>>>(resid, g2, Abf, nullptr);
  gemm_resid_kernel<<<ggrid, 256, GEMM_LDS_BYTES, stream>>>(Abf, Wt2, resid);
  addnorm_kernel<true><<<Tdim, 256, 0, stream>>>(resid, g3, nullptr, out);
}
